// MultiQuerySelfAttention_15264313770252
// MI455X (gfx1250) — matrix operand order repaired and checked by reading, not compiled, not run on hardware
//
#include <hip/hip_runtime.h>
#include <hip/hip_bf16.h>
#include <math.h>

#define B_   2
#define S_   2048
#define E_   2048
#define H_   16
#define HD_  128

typedef __attribute__((ext_vector_type(16))) _Float16 v16h;
typedef __attribute__((ext_vector_type(8)))  float    v8f;

union Frag { v16h h; uint4 q[2]; };

// ---------------------------------------------------------------------------
// A-matrix fragment (16x32 f16, MxK). Caller passes pointer to this lane's row
// (row = lane & 15), 16B-aligned, K contiguous.
// lanes 0-15: K 0-7 & 16-23 ; lanes 16-31: K 8-15 & 24-31.
__device__ inline v16h load_a_frag(const _Float16* rowp, int lane) {
    const int ko = (lane < 16) ? 0 : 8;
    Frag f;
    f.q[0] = *(const uint4*)(rowp + ko);
    f.q[1] = *(const uint4*)(rowp + ko + 16);
    return f.h;
}

// B-matrix fragment (32x16 f16, KxN) from Bt[N][K] row-major (K contiguous).
// lanes 0-15: K 0-7 & 16-23 ; lanes 16-31: K 8-15 & 24-31.
__device__ inline v16h load_b_frag(const _Float16* rowp, int lane) {
    const int ko = (lane < 16) ? 0 : 8;
    Frag f;
    f.q[0] = *(const uint4*)(rowp + ko);
    f.q[1] = *(const uint4*)(rowp + ko + 16);
    return f.h;
}

__device__ inline v8f wmma_f16(v16h a, v16h b, v8f c) {
    return __builtin_amdgcn_wmma_f32_16x16x32_f16(false, a, false, b,
                                                  (short)0, c, false, false);
}

// Async global->LDS 16B copy (CDNA5 GLOBAL_LOAD_ASYNC_TO_LDS_B128, ASYNCcnt).
// ldsoff = wave-relative LDS byte address (low 32 bits of generic address).
__device__ inline void async_copy_b128(unsigned ldsoff, const _Float16* gp) {
    unsigned long long ga = (unsigned long long)(uintptr_t)gp;
    asm volatile("global_load_async_to_lds_b128 %0, %1, off"
                 :: "v"(ldsoff), "v"(ga) : "memory");
}

__device__ inline unsigned lds_addr(const void* p) {
    return (unsigned)(uintptr_t)p;   // LDS aperture: low 32 bits = LDS offset
}

// ---------------------------------------------------------------------------
// f32 -> f16 conversion (vectorized by 4)
__global__ __launch_bounds__(256) void cvt_f16_kernel(const float* __restrict__ x,
                                                      _Float16* __restrict__ xh, int n4) {
    int i = blockIdx.x * 256 + threadIdx.x;
    if (i < n4) {
        float4 v = ((const float4*)x)[i];
        _Float16* o = xh + (size_t)i * 4;
        o[0] = (_Float16)v.x; o[1] = (_Float16)v.y;
        o[2] = (_Float16)v.z; o[3] = (_Float16)v.w;
    }
}

// W[K,N] f32 -> WT[N,K] f16 (one-time pre-pass; removes cvt from GEMM hot loop)
__global__ __launch_bounds__(256) void transpose_w_kernel(const float* __restrict__ W,
                                                          _Float16* __restrict__ WT,
                                                          int K, int N) {
    int idx = blockIdx.x * 256 + threadIdx.x;       // over N*K
    int k = idx % K;
    int n = idx / K;
    WT[idx] = (_Float16)W[(size_t)k * N + n];
}

// v [B,S,Hd] f16 -> vT [B,Hd,S] f16
__global__ __launch_bounds__(256) void transpose_v_kernel(const _Float16* __restrict__ vh,
                                                          _Float16* __restrict__ vT) {
    int idx = blockIdx.x * 256 + threadIdx.x;       // B*S*Hd elements
    int d = idx % HD_;
    int s = (idx / HD_) % S_;
    int b = idx / (S_ * HD_);
    vT[((size_t)b * HD_ + d) * S_ + s] = vh[idx];
}

// ---------------------------------------------------------------------------
// Tiled GEMM: C[M,N] = A[M,K](f16) * Bt[N,K](f16, pre-transposed) + bias[N]
// 256 threads = 8 waves; 128x128 block tile; wave = 32(M) x 64(N) sub-tile.
// Tiles staged into LDS with async b128 copies (no VGPR round-trip).
template <bool OUT_F32>
__global__ __launch_bounds__(256) void gemm_bias_kernel(
    const _Float16* __restrict__ A, const _Float16* __restrict__ Bt,
    const float* __restrict__ bias, _Float16* __restrict__ Ch,
    float* __restrict__ Cf, int M, int N, int K) {

    __shared__ _Float16 As[128][40];   // [m][k], stride 80B (16B multiple)
    __shared__ _Float16 Bs[128][40];   // [n][k]

    const int tid  = threadIdx.x;
    const int wv   = tid >> 5;
    const int lane = tid & 31;
    const int m0 = blockIdx.y * 128;
    const int n0 = blockIdx.x * 128;
    const int msub = (wv & 3) * 32;    // 4 row groups of 32
    const int nsub = (wv >> 2) * 64;   // 2 col groups of 64
    const int rl   = lane & 15;

    v8f acc[2][4] = {};

    for (int k0 = 0; k0 < K; k0 += 32) {
        // ---- async stage A and Bt tiles (128 rows x 32 halfs = 512 16B segs each)
        #pragma unroll
        for (int j = 0; j < 2; j++) {
            int s   = tid * 2 + j;          // 0..511
            int row = s >> 2;               // 4 segments per row
            int col = (s & 3) * 8;          // in halfs
            async_copy_b128(lds_addr(&As[row][col]),
                            A  + (size_t)(m0 + row) * K + k0 + col);
            async_copy_b128(lds_addr(&Bs[row][col]),
                            Bt + (size_t)(n0 + row) * K + k0 + col);
        }
        asm volatile("s_wait_asynccnt 0" ::: "memory");
        __syncthreads();

        v16h afr[2], bfr[4];
        #pragma unroll
        for (int mt = 0; mt < 2; mt++)
            afr[mt] = load_a_frag(&As[msub + mt * 16 + rl][0], lane);
        #pragma unroll
        for (int nt = 0; nt < 4; nt++)
            bfr[nt] = load_b_frag(&Bs[nsub + nt * 16 + rl][0], lane);

        #pragma unroll
        for (int mt = 0; mt < 2; mt++)
            #pragma unroll
            for (int nt = 0; nt < 4; nt++)
                acc[mt][nt] = wmma_f16(afr[mt], bfr[nt], acc[mt][nt]);
        __syncthreads();
    }

    // ---- epilogue: bias + store (C/D layout: lane<16 rows 0-7, lane>=16 rows 8-15)
    const int rbase = (lane < 16) ? 0 : 8;
    #pragma unroll
    for (int nt = 0; nt < 4; nt++) {
        int n = n0 + nsub + nt * 16 + rl;
        float bv = bias[n];
        #pragma unroll
        for (int mt = 0; mt < 2; mt++) {
            #pragma unroll
            for (int r = 0; r < 8; r++) {
                int m = m0 + msub + mt * 16 + rbase + r;
                float v = acc[mt][nt][r] + bv;
                if (OUT_F32) Cf[(size_t)m * N + n] = v;
                else         Ch[(size_t)m * N + n] = (_Float16)v;
            }
        }
    }
}

// ---------------------------------------------------------------------------
// Flash attention (MQA): q[B,S,E] f16 (viewed [B,S,H,Hd]), k[B,S,Hd] f16,
// vT[B,Hd,S] f16 -> attn[B,S,E] f16.  Block = 128 q-rows, 8 waves x 16 rows.
__global__ __launch_bounds__(256) void attn_kernel(
    const _Float16* __restrict__ qh, const _Float16* __restrict__ kh,
    const _Float16* __restrict__ vT, _Float16* __restrict__ attnh) {

    __shared__ _Float16 Pl[8][16][72];   // per-wave prob tile, stride 144B

    const int tid  = threadIdx.x;
    const int wv   = tid >> 5;
    const int lane = tid & 31;
    const int rl   = lane & 15;
    const int rbase = (lane < 16) ? 0 : 8;

    const int bh = blockIdx.y;           // B*H
    const int b  = bh / H_;
    const int h  = bh % H_;
    const int qt = blockIdx.x;           // S/128
    const int qr0 = qt * 128 + wv * 16;  // this wave's first q row

    const float scale = 0.08838834764831845f;          // 1/sqrt(128)
    const float c = scale * 1.4426950408889634f;       // scale * log2(e)

    // Q fragments resident in VGPRs: 4 K-chunks over Hd=128
    v16h qf[4];
    {
        const _Float16* qrow = qh + ((size_t)b * S_ + qr0 + rl) * E_ + h * HD_;
        #pragma unroll
        for (int kc = 0; kc < 4; kc++)
            qf[kc] = load_a_frag(qrow + kc * 32, lane);
    }

    v8f oacc[8] = {};
    float mrow[8], lrow[8];
    #pragma unroll
    for (int r = 0; r < 8; r++) { mrow[r] = -INFINITY; lrow[r] = 0.0f; }

    for (int kt = 0; kt < S_ / 64; kt++) {
        const int k0 = kt * 64;

        // ---- scores: 4 key tiles of 16, K over Hd in 4 chunks of 32
        v8f sacc[4] = {};
        #pragma unroll
        for (int nt = 0; nt < 4; nt++) {
            const _Float16* krow = kh + ((size_t)b * S_ + k0 + nt * 16 + rl) * HD_;
            #pragma unroll
            for (int kc = 0; kc < 4; kc++) {
                v16h kf = load_b_frag(krow + kc * 32, lane);
                sacc[nt] = wmma_f16(qf[kc], kf, sacc[nt]);
            }
        }

        // ---- online softmax (rows striped across 16-lane halves)
        float pm[8], mnew[8], corr[8], psum[8];
        #pragma unroll
        for (int r = 0; r < 8; r++) {
            float v = sacc[0][r];
            v = fmaxf(v, sacc[1][r]); v = fmaxf(v, sacc[2][r]); v = fmaxf(v, sacc[3][r]);
            pm[r] = v;
        }
        #pragma unroll
        for (int r = 0; r < 8; r++) {
            #pragma unroll
            for (int off = 1; off < 16; off <<= 1)
                pm[r] = fmaxf(pm[r], __shfl_xor(pm[r], off, 32));
            mnew[r] = fmaxf(mrow[r], pm[r]);
            corr[r] = exp2f((mrow[r] - mnew[r]) * c);
            psum[r] = 0.0f;
        }
        #pragma unroll
        for (int nt = 0; nt < 4; nt++) {
            #pragma unroll
            for (int r = 0; r < 8; r++) {
                float p = exp2f((sacc[nt][r] - mnew[r]) * c);
                psum[r] += p;
                Pl[wv][rbase + r][nt * 16 + rl] = (_Float16)p;
            }
        }
        #pragma unroll
        for (int r = 0; r < 8; r++) {
            #pragma unroll
            for (int off = 1; off < 16; off <<= 1)
                psum[r] += __shfl_xor(psum[r], off, 32);
            lrow[r] = lrow[r] * corr[r] + psum[r];
            mrow[r] = mnew[r];
        }
        #pragma unroll
        for (int dt = 0; dt < 8; dt++)
            #pragma unroll
            for (int r = 0; r < 8; r++)
                oacc[dt][r] *= corr[r];

        // per-wave LDS tile: wait our own DS stores (in-order with later loads)
        asm volatile("s_wait_dscnt 0" ::: "memory");

        // ---- PV: P(16x64) as A over 2 K-chunks; vT rows give contiguous-K B frags
        #pragma unroll
        for (int kc = 0; kc < 2; kc++) {
            v16h pf = load_a_frag(&Pl[wv][rl][kc * 32], lane);
            #pragma unroll
            for (int dt = 0; dt < 8; dt++) {
                const _Float16* vp =
                    vT + ((size_t)b * HD_ + dt * 16 + rl) * S_ + k0 + kc * 32;
                v16h vf = load_b_frag(vp, lane);
                oacc[dt] = wmma_f16(pf, vf, oacc[dt]);
            }
        }
    }

    // ---- normalize + store f16 attn output at [b, qrow, h*Hd + d]
    #pragma unroll
    for (int dt = 0; dt < 8; dt++) {
        #pragma unroll
        for (int r = 0; r < 8; r++) {
            float ov = oacc[dt][r] / lrow[r];
            int qrow = qr0 + rbase + r;
            attnh[((size_t)b * S_ + qrow) * E_ + h * HD_ + dt * 16 + rl] = (_Float16)ov;
        }
    }
}

// ---------------------------------------------------------------------------
extern "C" void kernel_launch(void* const* d_in, const int* in_sizes, int n_in,
                              void* d_out, int out_size, void* d_ws, size_t ws_size,
                              hipStream_t stream) {
    const float* x  = (const float*)d_in[0];
    const float* Wq = (const float*)d_in[1];
    const float* bq = (const float*)d_in[2];
    const float* Wk = (const float*)d_in[3];
    const float* bk = (const float*)d_in[4];
    const float* Wv = (const float*)d_in[5];
    const float* bv = (const float*)d_in[6];
    const float* Wo = (const float*)d_in[7];
    const float* bo = (const float*)d_in[8];
    float* out = (float*)d_out;

    const size_t MB = 1024u * 1024u;
    char* ws = (char*)d_ws;
    _Float16* xh   = (_Float16*)(ws + 0 * MB);    // 16 MB  x as f16
    _Float16* qh   = (_Float16*)(ws + 16 * MB);   // 16 MB  q
    _Float16* kh   = (_Float16*)(ws + 32 * MB);   //  1 MB  k
    _Float16* vh   = (_Float16*)(ws + 33 * MB);   //  1 MB  v
    _Float16* vTh  = (_Float16*)(ws + 34 * MB);   //  1 MB  v^T
    _Float16* atth = (_Float16*)(ws + 35 * MB);   // 16 MB  attn (pre O-proj)
    _Float16* WqT  = (_Float16*)(ws + 51 * MB);   //  8 MB  Wq^T f16
    _Float16* WoT  = (_Float16*)(ws + 59 * MB);   //  8 MB  Wo^T f16
    _Float16* WkT  = (_Float16*)(ws + 67 * MB);   // .5 MB  Wk^T f16
    _Float16* WvT  = (_Float16*)(ws + 67 * MB + 512 * 1024);  // .5 MB

    const int M = B_ * S_;                        // 4096

    // 1) x -> f16
    {
        int n4 = (B_ * S_ * E_) / 4;
        cvt_f16_kernel<<<(n4 + 255) / 256, 256, 0, stream>>>(x, xh, n4);
    }
    // 2) weights -> transposed f16 (one-time)
    transpose_w_kernel<<<(E_ * E_) / 256, 256, 0, stream>>>(Wq, WqT, E_, E_);
    transpose_w_kernel<<<(E_ * HD_) / 256, 256, 0, stream>>>(Wk, WkT, E_, HD_);
    transpose_w_kernel<<<(E_ * HD_) / 256, 256, 0, stream>>>(Wv, WvT, E_, HD_);
    transpose_w_kernel<<<(E_ * E_) / 256, 256, 0, stream>>>(Wo, WoT, E_, E_);
    // 3-5) projections
    gemm_bias_kernel<false><<<dim3(E_ / 128, M / 128), 256, 0, stream>>>(
        xh, WqT, bq, qh, nullptr, M, E_, E_);
    gemm_bias_kernel<false><<<dim3(HD_ / 128, M / 128), 256, 0, stream>>>(
        xh, WkT, bk, kh, nullptr, M, HD_, E_);
    gemm_bias_kernel<false><<<dim3(HD_ / 128, M / 128), 256, 0, stream>>>(
        xh, WvT, bv, vh, nullptr, M, HD_, E_);
    // 6) v -> v^T
    transpose_v_kernel<<<(B_ * S_ * HD_) / 256, 256, 0, stream>>>(vh, vTh);
    // 7) fused flash attention
    attn_kernel<<<dim3(S_ / 128, B_ * H_), 256, 0, stream>>>(qh, kh, vTh, atth);
    // 8) output projection (f32 out)
    gemm_bias_kernel<true><<<dim3(E_ / 128, M / 128), 256, 0, stream>>>(
        atth, WoT, bo, nullptr, out, M, E_, E_);
}
